// CRMF_8366596293120
// MI455X (gfx1250) — matrix operand order repaired and checked by reading, not compiled, not run on hardware
//
#include <hip/hip_runtime.h>
#include <hip/hip_bf16.h>

// GAT forward on MI455X (gfx1250, wave32).
// Outputs (concatenated in d_out): output [8,4,2048,64] f32, attn [8,4,2048,2048] f32.

typedef _Float16 v16h __attribute__((ext_vector_type(16)));
typedef float    v8f  __attribute__((ext_vector_type(8)));
typedef float    v4f  __attribute__((ext_vector_type(4)));

#define BS    8
#define NH    4
#define NN    2048
#define FIN   32
#define FOUT  64
#define TM    16        // rows per stage-2 block
#define PSTR  2052      // padded LDS row stride (floats) to spread banks

// ---------------- Stage 1: h_prime = h @ w[h]; src/dst dots; h_primeT in f16 ----
__global__ __launch_bounds__(256)
void gat_stage1(const float* __restrict__ h, const float* __restrict__ w,
                const float* __restrict__ a_src, const float* __restrict__ a_dst,
                _Float16* __restrict__ hpT, float* __restrict__ srcv,
                float* __restrict__ dstv) {
    extern __shared__ float sm1[];
    float* sw  = sm1;            // [32*64]
    float* sas = sm1 + FIN*FOUT; // [64]
    float* sad = sas + FOUT;     // [64]

    const int tid   = threadIdx.x;
    const int chunk = blockIdx.x & 7;     // 2048/256 row chunks
    const int bh    = blockIdx.x >> 3;    // b*NH + h
    const int b     = bh >> 2;
    const int hh    = bh & 3;

    for (int i = tid; i < FIN*FOUT; i += 256) sw[i] = w[hh*FIN*FOUT + i];
    if (tid < FOUT) { sas[tid] = a_src[hh*FOUT + tid]; sad[tid] = a_dst[hh*FOUT + tid]; }
    __syncthreads();

    const int row = chunk*256 + tid;
    float hr[FIN];
    const float* hrow = h + ((size_t)b*NN + row)*FIN;
#pragma unroll
    for (int f = 0; f < FIN; ++f) hr[f] = hrow[f];

    float sdot = 0.f, ddot = 0.f;
    _Float16* outp = hpT + (size_t)bh*FOUT*NN + row;   // [bh][o][k=row]
    for (int o0 = 0; o0 < FOUT; o0 += 8) {
        float acc[8] = {0.f,0.f,0.f,0.f,0.f,0.f,0.f,0.f};
#pragma unroll
        for (int f = 0; f < FIN; ++f) {
            const float hv = hr[f];
            const float* wrow = sw + f*FOUT + o0;
#pragma unroll
            for (int oo = 0; oo < 8; ++oo) acc[oo] = fmaf(hv, wrow[oo], acc[oo]);
        }
#pragma unroll
        for (int oo = 0; oo < 8; ++oo) {
            const float v = acc[oo];
            sdot += v * sas[o0 + oo];
            ddot += v * sad[o0 + oo];
            outp[(size_t)(o0 + oo) * NN] = (_Float16)v;
        }
    }
    srcv[(size_t)bh*NN + row] = sdot;
    dstv[(size_t)bh*NN + row] = ddot;
}

// ---------------- Stage 2: fused softmax + attn write + attn@h_prime (WMMA) ----
__global__ __launch_bounds__(256)
void gat_stage2(const _Float16* __restrict__ hpT, const float* __restrict__ srcg,
                const float* __restrict__ dstg, const float* __restrict__ bias,
                float* __restrict__ out, float* __restrict__ attn) {
    extern __shared__ float sm[];
    float* P     = sm;                 // [TM][PSTR]  unnormalized probs (exp)
    float* dsh   = P + TM*PSTR;        // [NN]
    float* cred  = dsh + NN;           // [4*256] partial C reduction
    float* srcsh = cred + 4*256;       // [16]
    float* msh   = srcsh + 16;         // [16]
    float* rsum  = msh + 16;           // [16]
    float* invsh = rsum + 16;          // [16]
    float* wred  = invsh + 16;         // [9]

    const int tid  = threadIdx.x;
    const int lane = tid & 31;
    const int wv   = tid >> 5;
    const int rt   = blockIdx.x & 127;
    const int bh   = blockIdx.x >> 7;
    const int row0 = rt * TM;

    // ---- load dst row into LDS; block max of dst ----
    const float* dstrow = dstg + (size_t)bh*NN;
    float vmax = -3.0e38f;
    for (int j = tid; j < NN; j += 256) {
        const float d = dstrow[j];
        dsh[j] = d;
        vmax = fmaxf(vmax, d);
    }
#pragma unroll
    for (int off = 16; off; off >>= 1) vmax = fmaxf(vmax, __shfl_xor(vmax, off, 32));
    if (lane == 0) wred[wv] = vmax;
    if (tid < TM) { srcsh[tid] = srcg[(size_t)bh*NN + row0 + tid]; rsum[tid] = 0.f; }
    __syncthreads();
    if (tid == 0) {
        float m = wred[0];
#pragma unroll
        for (int k = 1; k < 8; ++k) m = fmaxf(m, wred[k]);
        wred[8] = m;
    }
    __syncthreads();
    if (tid < TM) {  // row max = lrelu(src_i + max_j dst_j), lrelu monotone
        const float x = srcsh[tid] + wred[8];
        msh[tid] = x > 0.f ? x : 0.2f * x;
    }
    __syncthreads();

    // ---- pass A: p_ij = exp(lrelu(src_i + dst_j) - m_i); row sums ----
    float accs[TM];
#pragma unroll
    for (int i = 0; i < TM; ++i) accs[i] = 0.f;
    for (int q = 0; q < 8; ++q) {
        const int j = tid + 256*q;
        const float d = dsh[j];
#pragma unroll
        for (int i = 0; i < TM; ++i) {
            float x = srcsh[i] + d;
            x = x > 0.f ? x : 0.2f * x;
            const float p = __expf(x - msh[i]);
            accs[i] += p;
            P[i*PSTR + j] = p;
        }
    }
#pragma unroll
    for (int i = 0; i < TM; ++i) atomicAdd(&rsum[i], accs[i]);  // ds_add_f32
    __syncthreads();
    if (tid < TM) invsh[tid] = 1.0f / rsum[tid];
    __syncthreads();

    // ---- stream normalized attn tile to global (non-temporal: write-once,
    //      keep L2 for the reused hpT tiles). P stays UNNORMALIZED in LDS. ----
    float* attnrow = attn + ((size_t)bh*NN + row0)*NN;
#pragma unroll 4
    for (int i = 0; i < TM; ++i) {
        const float inv = invsh[i];
        const float* pr = P + i*PSTR + tid*8;
        v4f v0 = *(const v4f*)pr * inv;
        v4f v1 = *(const v4f*)(pr + 4) * inv;
        float* gp = attnrow + (size_t)i*NN + tid*8;
        __builtin_nontemporal_store(v0, (v4f*)gp);
        __builtin_nontemporal_store(v1, (v4f*)(gp + 4));
    }
    // no barrier needed: P is read-only from here on

    // ---- GEMM on unnormalized P: C = P @ hpT^T, then scale rows by inv[M].
    //      v_wmma_f32_16x16x32_f16; K=2048 split across wave pairs. ----
    const int ntile = wv & 3;      // which 16-col tile of FOUT
    const int khalf = wv >> 2;     // K split across wave pairs
    const int r  = lane & 15;
    const int hi = lane >> 4;
    const int ocol = ntile*16 + r;
    const _Float16* bbase = hpT + ((size_t)bh*FOUT + ocol)*NN + khalf*1024 + hi*16;
    const float*    abase = P + r*PSTR + khalf*1024 + hi*16;

    v8f acc = {0.f,0.f,0.f,0.f,0.f,0.f,0.f,0.f};
    for (int kc = 0; kc < 32; ++kc) {
        const float* ap = abase + kc*32;
        const v4f a0 = *(const v4f*)(ap);
        const v4f a1 = *(const v4f*)(ap + 4);
        const v4f a2 = *(const v4f*)(ap + 8);
        const v4f a3 = *(const v4f*)(ap + 12);
        v16h A;  // A layout: lane {M=l&15}; elems 0-15: K=(l>>4)*16+e (matches B)
        A[0]=(_Float16)a0[0];  A[1]=(_Float16)a0[1];  A[2]=(_Float16)a0[2];  A[3]=(_Float16)a0[3];
        A[4]=(_Float16)a1[0];  A[5]=(_Float16)a1[1];  A[6]=(_Float16)a1[2];  A[7]=(_Float16)a1[3];
        A[8]=(_Float16)a2[0];  A[9]=(_Float16)a2[1];  A[10]=(_Float16)a2[2]; A[11]=(_Float16)a2[3];
        A[12]=(_Float16)a3[0]; A[13]=(_Float16)a3[1]; A[14]=(_Float16)a3[2]; A[15]=(_Float16)a3[3];
        const v16h B = *(const v16h*)(bbase + kc*32);  // B: N=l&15, K=(l>>4)*16+e, contiguous 32B
        acc = __builtin_amdgcn_wmma_f32_16x16x32_f16(false, A, false, B, (short)0, acc, false, false);
    }

    if (khalf == 1) {
#pragma unroll
        for (int v = 0; v < 8; ++v) cred[ntile*256 + v*32 + lane] = acc[v];
    }
    __syncthreads();
    if (khalf == 0) {
        const int ncol = ntile*16 + (lane & 15);
        const float bb = bias[ncol];
        const size_t obase = ((size_t)bh*NN + row0) * FOUT;
#pragma unroll
        for (int v = 0; v < 8; ++v) {
            const int M = v + 8*hi;            // C/D layout: M=v+8*(l>>4), N=l&15
            const float cv = (acc[v] + cred[ntile*256 + v*32 + lane]) * invsh[M];
            out[obase + (size_t)M*FOUT + ncol] = cv + bb;
        }
    }
}

extern "C" void kernel_launch(void* const* d_in, const int* in_sizes, int n_in,
                              void* d_out, int out_size, void* d_ws, size_t ws_size,
                              hipStream_t stream) {
    const float* h     = (const float*)d_in[0];
    const float* w     = (const float*)d_in[1];
    const float* a_src = (const float*)d_in[2];
    const float* a_dst = (const float*)d_in[3];
    const float* bvec  = (const float*)d_in[4];

    float* out  = (float*)d_out;                                   // [8,4,2048,64]
    float* attn = out + (size_t)BS*NH*NN*FOUT;                     // [8,4,2048,2048]

    // workspace: hpT f16 [32][64][2048] (8 MB), src f32 [32][2048], dst f32 [32][2048]
    _Float16* hpT = (_Float16*)d_ws;
    float* srcv = (float*)((char*)d_ws + (size_t)BS*NH*FOUT*NN*sizeof(_Float16));
    float* dstv = srcv + (size_t)BS*NH*NN;

    const size_t sm1 = (FIN*FOUT + 2*FOUT) * sizeof(float);
    gat_stage1<<<BS*NH*8, 256, sm1, stream>>>(h, w, a_src, a_dst, hpT, srcv, dstv);

    const size_t sm2 = (TM*PSTR + NN + 4*256 + 4*16 + 9) * sizeof(float);
    gat_stage2<<<BS*NH*(NN/TM), 256, sm2, stream>>>(hpT, srcv, dstv, bvec, out, attn);
}
